// ScaledDotProductAttention_86955907875523
// MI455X (gfx1250) — matrix operand order repaired and checked by reading, not compiled, not run on hardware
//
#include <hip/hip_runtime.h>
#include <hip/hip_bf16.h>

typedef __attribute__((ext_vector_type(16))) _Float16 v16h;
typedef __attribute__((ext_vector_type(8)))  float    v8f;

#define BB   4
#define HHN  8
#define SSL  2048
#define DKK  64
#define SSTR 2052   // padded LDS row stride in floats (2052 % 64 == 4 -> conflict-free)

// One workgroup = one (b, h, 16-row q tile). 256 threads = 8 wave32.
__global__ __launch_bounds__(256) void attn_fused_kernel(
    const float* __restrict__ Q, const float* __restrict__ K,
    const float* __restrict__ V, const unsigned char* __restrict__ mask,
    const float* __restrict__ edge,
    float* __restrict__ ctx, float* __restrict__ attn, float* __restrict__ scores)
{
    extern __shared__ float smem[];
    float* sS   = smem;               // 16 * SSTR floats: masked scores, then exp(p)
    float* sInv = smem + 16 * SSTR;   // 16 floats: 1/rowsum

    const int tid   = threadIdx.x;
    const int wave  = tid >> 5;
    const int lane  = tid & 31;
    const int laneN = lane & 15;
    const int hi    = lane >> 4;      // 0 = low half, 1 = high half of wave

    const int  qt = blockIdx.x;       // 0..127
    const int  h  = blockIdx.y;
    const int  b  = blockIdx.z;
    const int  q0 = qt * 16;
    const long bh = (long)b * HHN + h;

    const float* Qb = Q + bh * SSL * DKK;
    const float* Kb = K + bh * SSL * DKK;
    const float* Vb = V + bh * SSL * DKK;
    const unsigned char* Mb = mask + bh * (long)SSL * SSL;   // bool stored as 1 byte
    const float* Eb = edge + (long)b * SSL * SSL;
    float* scoresB = scores + bh * (long)SSL * SSL;
    float* attnB   = attn   + bh * (long)SSL * SSL;
    float* ctxB    = ctx    + bh * SSL * DKK;

    // ---- Q tile as two A-fragments (16x32 f16 chunks over DK=64) ------------
    // ISA 16-bit A 16x32 layout: lane m=lane&15; elements e=0..7 -> K = kb + hofs + e,
    // e=8..15 -> K = kb + 16 + hofs + (e-8), hofs = (lane>=16) ? 8 : 0.
    v16h aq[2];
    {
        const float* qrow = Qb + (long)(q0 + laneN) * DKK;
        const int hofs = hi * 8;
#pragma unroll
        for (int c = 0; c < 2; ++c) {
            const int kb = c * 32;
#pragma unroll
            for (int j = 0; j < 8; ++j) {
                aq[c][j]     = (_Float16)qrow[kb + hofs + j];
                aq[c][8 + j] = (_Float16)qrow[kb + 16 + hofs + j];
            }
        }
    }

    const float scale = 0.125f;   // 1/sqrt(64)

    // ---- Phase 1: scores tile = mask(QK^T * scale + edge) -------------------
    // B-fragment (32x16): lane n=lane&15 holds K-features of K-row (kt*16+n) in
    // table order: e<8 -> hofs+e, e>=8 -> 16+hofs+(e-8), hofs = 8 for high half.
    for (int kt = wave; kt < SSL / 16; kt += 8) {
        v8f acc = {};
#pragma unroll
        for (int c = 0; c < 2; ++c) {
            const int kb = c * 32;
            const float* krow = Kb + (long)(kt * 16 + laneN) * DKK + kb + hi * 8;
            v16h bk;
#pragma unroll
            for (int j = 0; j < 16; ++j) bk[j] = (_Float16)krow[j + (j >= 8 ? 8 : 0)];
            acc = __builtin_amdgcn_wmma_f32_16x16x32_f16(
                false, aq[c], false, bk, (short)0, acc, false, false);
        }
        // D layout: reg r -> row m = r + 8*hi, col = kt*16 + (lane&15)
#pragma unroll
        for (int r = 0; r < 8; ++r) {
            const int  m   = r + hi * 8;
            const int  col = kt * 16 + laneN;
            const long row = q0 + m;
            float s = acc[r] * scale + Eb[row * SSL + col];
            if (Mb[row * SSL + col]) s = -1e9f;
            scoresB[row * SSL + col] = s;
            sS[m * SSTR + col]       = s;
        }
    }
    __syncthreads();

    // ---- Phase 2: row softmax (wave w owns rows 2w, 2w+1), keep p in LDS ----
#pragma unroll
    for (int rr = 0; rr < 2; ++rr) {
        const int m = wave * 2 + rr;
        float* rowp = sS + m * SSTR;
        float mx = -3.4e38f;
        for (int col = lane; col < SSL; col += 32) mx = fmaxf(mx, rowp[col]);
#pragma unroll
        for (int o = 16; o > 0; o >>= 1) mx = fmaxf(mx, __shfl_xor(mx, o, 32));
        float sum = 0.f;
        for (int col = lane; col < SSL; col += 32) {
            float p = __expf(rowp[col] - mx);
            rowp[col] = p;
            sum += p;
        }
#pragma unroll
        for (int o = 16; o > 0; o >>= 1) sum += __shfl_xor(sum, o, 32);
        if (lane == 0) sInv[m] = 1.0f / sum;
    }
    __syncthreads();

    // ---- Phase 3: stream normalized attention out (coalesced) ---------------
    for (int idx = tid; idx < 16 * SSL; idx += 256) {
        const int m   = idx >> 11;          // / 2048
        const int col = idx & (SSL - 1);
        attnB[(long)(q0 + m) * SSL + col] = sS[m * SSTR + col] * sInv[m];
    }
    __syncthreads();

    // ---- Phase 4: context = P @ V, waves 0..3 own one 16-col output tile ----
    if (wave < 4) {
        const int n0   = wave * 16 + laneN;
        const int hofs = hi * 8;
        v8f acc = {};
        for (int kb = 0; kb < SSL; kb += 32) {
            v16h ap, bv;
#pragma unroll
            for (int j = 0; j < 8; ++j) {
                ap[j]     = (_Float16)sS[laneN * SSTR + kb + hofs + j];
                ap[8 + j] = (_Float16)sS[laneN * SSTR + kb + 16 + hofs + j];
            }
            const float* vcol = Vb + (long)(kb + hi * 8) * DKK + n0;
#pragma unroll
            for (int j = 0; j < 16; ++j) bv[j] = (_Float16)vcol[(long)(j + (j >= 8 ? 8 : 0)) * DKK];
            acc = __builtin_amdgcn_wmma_f32_16x16x32_f16(
                false, ap, false, bv, (short)0, acc, false, false);
        }
#pragma unroll
        for (int r = 0; r < 8; ++r) {
            const int m = r + hi * 8;
            ctxB[(long)(q0 + m) * DKK + n0] = acc[r] * sInv[m];
        }
    }
}

extern "C" void kernel_launch(void* const* d_in, const int* in_sizes, int n_in,
                              void* d_out, int out_size, void* d_ws, size_t ws_size,
                              hipStream_t stream) {
    const float*         Q    = (const float*)d_in[0];
    const float*         K    = (const float*)d_in[1];
    const float*         V    = (const float*)d_in[2];
    const unsigned char* mask = (const unsigned char*)d_in[3]; // jnp.bool_ -> 1 byte
    const float*         edge = (const float*)d_in[4];

    // d_out = context | attn | scores (reference return order), all f32
    float* ctx    = (float*)d_out;
    float* attn   = ctx  + (long)BB * HHN * SSL * DKK;
    float* scores = attn + (long)BB * HHN * SSL * SSL;

    const size_t shbytes = (size_t)(16 * SSTR + 16) * sizeof(float); // ~131 KB (<320 KB WGP LDS)
    (void)hipFuncSetAttribute((const void*)attn_fused_kernel,
                              hipFuncAttributeMaxDynamicSharedMemorySize, (int)shbytes);

    dim3 grid(SSL / 16, HHN, BB);   // (128, 8, 4)
    dim3 block(256);
    attn_fused_kernel<<<grid, block, shbytes, stream>>>(Q, K, V, mask, edge,
                                                        ctx, attn, scores);
}
